// MultiHeadAttention_20083267076660
// MI455X (gfx1250) — matrix operand order repaired and checked by reading, not compiled, not run on hardware
//
#include <hip/hip_runtime.h>
#include <hip/hip_bf16.h>
#include <math.h>

typedef __bf16 bf16_t;
typedef __attribute__((ext_vector_type(16))) bf16_t v16bf;
typedef __attribute__((ext_vector_type(8)))  float  v8f;
typedef __attribute__((ext_vector_type(4)))  int    v4i;
typedef __attribute__((ext_vector_type(8)))  int    v8i;
typedef __attribute__((ext_vector_type(4)))  unsigned v4u;

#define B_   2
#define S_   2048
#define D_   2048
#define H_   16
#define HD_  128
#define N3D_ 6144
#define BS_  (B_ * S_)   // 4096

// ---------------- GEMM tiling ----------------
#define BM  128
#define BN  128
#define KC  32
#define LDK 40            // KC + 8 halves pad (80B rows, 16B aligned) == TDM pad 4dw/16dw
#define BUFH (BM * LDK)   // halves per LDS buffer

// ---------------- Attention tiling ----------------
#define KB_KEYS 64
#define KT_LD   136   // 128 hd + 8 pad
#define VT_LD   72    // 64 keys + 8 pad
#define P_LD    72

#define WMMA_BF16(a, b, c) \
  __builtin_amdgcn_wmma_f32_16x16x32_bf16(false, (a), false, (b), (short)0, (c), false, false)

// ---------------- CDNA5 async global->LDS copy (ASYNCcnt tracked) ----------------
__device__ __forceinline__ void async_copy_b128(const bf16_t* g, bf16_t* l) {
#if __has_builtin(__builtin_amdgcn_global_load_async_to_lds_b128)
  __builtin_amdgcn_global_load_async_to_lds_b128(
      (__attribute__((address_space(1))) v4i*)g,
      (__attribute__((address_space(3))) v4i*)l, 0, 0);
#else
  unsigned la = (unsigned)(unsigned long long)(__attribute__((address_space(3))) void*)l;
  asm volatile("global_load_async_to_lds_b128 %0, %1, off"
               :: "v"(la), "v"((unsigned long long)g) : "memory");
#endif
}

#if __has_builtin(__builtin_amdgcn_s_wait_asynccnt)
#define WAIT_ASYNC(n) __builtin_amdgcn_s_wait_asynccnt(n)
#else
#define WAIT_ASYNC(n) asm volatile("s_wait_asynccnt %0" :: "i"(n) : "memory")
#endif

#if __has_builtin(__builtin_amdgcn_s_wait_tensorcnt)
#define WAIT_TENSOR(n) __builtin_amdgcn_s_wait_tensorcnt(n)
#else
#define WAIT_TENSOR(n) asm volatile("s_wait_tensorcnt %0" :: "i"(n) : "memory")
#endif

#if __has_builtin(__builtin_amdgcn_tensor_load_to_lds)
#define HAVE_TDM 1
#else
#define HAVE_TDM 0
#endif

#if HAVE_TDM
// TDM 2D tile load: tile_dim0=KC(32) halves, tile_dim1=128 rows, 2-byte elements,
// LDS padding 4 DWORDs every 16 DWORDs -> row stride LDK=40 halves.
__device__ __forceinline__ void tdm_load_2d(const bf16_t* gbase, unsigned lds_off,
                                            unsigned k_elems /*tensor_dim0 & stride*/,
                                            unsigned rows    /*tensor_dim1*/) {
  unsigned long long ga = (unsigned long long)gbase;
  v4u g0;
  g0[0] = 1u;                                   // count=1, user descriptor
  g0[1] = lds_off;                              // lds_addr (bytes)
  g0[2] = (unsigned)ga;                         // global_addr[31:0]
  g0[3] = (unsigned)(ga >> 32) | (2u << 30);    // global_addr[56:32] | type=2 (image)
  v8i g1;
  g1[0] = (int)((1u << 16)                      // data_size = 2 bytes
              | (1u << 20)                      // pad_enable
              | (3u << 22)                      // pad_interval: 16 DWORDs (one 64B row)
              | (3u << 25));                    // pad_amount: 4 DWORDs (16B)
  g1[1] = (int)(k_elems << 16);                 // tensor_dim0[15:0]
  g1[2] = (int)((k_elems >> 16) | (rows << 16)); // tensor_dim0[31:16] | tensor_dim1[15:0]
  g1[3] = (int)((rows >> 16) | ((unsigned)KC << 16)); // tensor_dim1[31:16] | tile_dim0=32
  g1[4] = BM;                                   // tile_dim1=128 | tile_dim2=0
  g1[5] = (int)k_elems;                         // tensor_dim0_stride[31:0] (elements)
  g1[6] = 0;
  g1[7] = 0;
  v4i zz = {};
#if defined(__clang_major__) && (__clang_major__ >= 23)
  v8i z8 = {};
  __builtin_amdgcn_tensor_load_to_lds(g0, g1, zz, zz, z8, 0);
#else
  __builtin_amdgcn_tensor_load_to_lds(g0, g1, zz, zz, 0);
#endif
}
#endif

// A-matrix fragment (16xK=32, bf16): lane l -> row l&15, halves (l>>4)*8+[0,8) and +16
__device__ __forceinline__ v16bf ld_frag_a(const bf16_t* base, int strideH, int lane) {
  int r = lane & 15, g = lane >> 4;
  const bf16_t* p = base + r * strideH + g * 8;
  union { v16bf v; uint4 q[2]; } u;
  u.q[0] = *(const uint4*)(p);
  u.q[1] = *(const uint4*)(p + 16);
  return u.v;
}

// B-matrix fragment (K=32 x 16, bf16): lane l -> col l&15, halves (l>>4)*8+[0,8) and +16
__device__ __forceinline__ v16bf ld_frag_b(const bf16_t* base, int strideH, int lane) {
  int c = lane & 15, g = lane >> 4;
  const bf16_t* p = base + c * strideH + g * 8;
  union { v16bf v; uint4 q[2]; } u;
  u.q[0] = *(const uint4*)(p);
  u.q[1] = *(const uint4*)(p + 16);
  return u.v;
}

// ---------------- fp32 -> bf16 cast ----------------
__global__ void cast_f32_to_bf16(const float* __restrict__ src,
                                 bf16_t* __restrict__ dst, int n) {
  int i = (blockIdx.x * blockDim.x + threadIdx.x) * 4;
  if (i + 3 < n) {
    float4 f = *(const float4*)(src + i);
    union { bf16_t h[4]; uint2 u; } o;
    o.h[0] = (bf16_t)f.x; o.h[1] = (bf16_t)f.y;
    o.h[2] = (bf16_t)f.z; o.h[3] = (bf16_t)f.w;
    *(uint2*)(dst + i) = o.u;
  }
}

// ------- shared GEMM mainloop: C(128x128) = A(M,K) @ W(N,K)^T -------
// TDM-driven double-buffered staging (wave 0 issues tensor DMAs); falls back to
// per-thread async global->LDS copies when TDM builtin is unavailable.
__device__ __forceinline__ void gemm_mainloop(const bf16_t* __restrict__ A,
                                              const bf16_t* __restrict__ W, int K,
                                              int m_block, int n_block,
                                              bf16_t* lA, bf16_t* lB, v8f acc[2][4]) {
  const int tid  = threadIdx.x;
  const int lane = tid & 31, wid = tid >> 5;
  const int wm = wid & 3, wn = wid >> 2;

  v8f z = {};
  for (int i = 0; i < 2; i++)
    for (int t = 0; t < 4; t++) acc[i][t] = z;

  const int niter = K / KC;

#if HAVE_TDM
  const unsigned ldsA = (unsigned)(size_t)lA;   // low 32 bits of generic ptr = LDS offset
  const unsigned ldsB = (unsigned)(size_t)lB;
  const bf16_t* gA0 = A + (size_t)m_block * K;
  const bf16_t* gB0 = W + (size_t)n_block * K;
  if (wid == 0) {   // one wave drives the Tensor Data Mover
    tdm_load_2d(gA0, ldsA, (unsigned)K, BM);
    tdm_load_2d(gB0, ldsB, (unsigned)K, BN);
  }
#else
  const int row = tid >> 1, ch = (tid & 1) << 4;
  const bf16_t* gA = A + (size_t)(m_block + row) * K + ch;
  const bf16_t* gB = W + (size_t)(n_block + row) * K + ch;
  bf16_t* sA = lA + row * LDK + ch;
  bf16_t* sB = lB + row * LDK + ch;
  async_copy_b128(gA, sA);
  async_copy_b128(gA + 8, sA + 8);
  async_copy_b128(gB, sB);
  async_copy_b128(gB + 8, sB + 8);
#endif

  for (int i = 0; i < niter; ++i) {
    const int cur = i & 1, nxt = cur ^ 1;
    __syncthreads();   // all waves done computing from buffer `nxt` (prev iter)

#if HAVE_TDM
    if (wid == 0) {
      if (i + 1 < niter) {
        tdm_load_2d(gA0 + (size_t)(i + 1) * KC, ldsA + (unsigned)nxt * (BUFH * 2), (unsigned)K, BM);
        tdm_load_2d(gB0 + (size_t)(i + 1) * KC, ldsB + (unsigned)nxt * (BUFH * 2), (unsigned)K, BN);
        WAIT_TENSOR(2);   // in-order per wave: current tile's 2 DMAs complete
      } else {
        WAIT_TENSOR(0);
      }
    }
#else
    if (i + 1 < niter) {
      const bf16_t* nA = gA + (size_t)(i + 1) * KC;
      const bf16_t* nB = gB + (size_t)(i + 1) * KC;
      async_copy_b128(nA,     sA + nxt * BUFH);
      async_copy_b128(nA + 8, sA + nxt * BUFH + 8);
      async_copy_b128(nB,     sB + nxt * BUFH);
      async_copy_b128(nB + 8, sB + nxt * BUFH + 8);
      WAIT_ASYNC(4);
    } else {
      WAIT_ASYNC(0);
    }
#endif
    __syncthreads();   // tile i resident for all waves

    const bf16_t* bA = lA + cur * BUFH;
    const bf16_t* bB = lB + cur * BUFH;
    v16bf af[2], bfr[4];
    for (int m = 0; m < 2; m++)
      af[m] = ld_frag_a(bA + (wm * 32 + 16 * m) * LDK, LDK, lane);
    for (int t = 0; t < 4; t++)
      bfr[t] = ld_frag_b(bB + (wn * 64 + 16 * t) * LDK, LDK, lane);
    for (int m = 0; m < 2; m++)
      for (int t = 0; t < 4; t++)
        acc[m][t] = WMMA_BF16(af[m], bfr[t], acc[m][t]);
  }
}

// ---------------- QKV projection: qkv = hs @ Wqkv^T + b, scattered per head ----------------
__global__ __launch_bounds__(256) void qkv_gemm_kernel(
    const bf16_t* __restrict__ hsB, const bf16_t* __restrict__ wB,
    const float* __restrict__ bias,
    bf16_t* __restrict__ qB, bf16_t* __restrict__ kB, bf16_t* __restrict__ vB) {
  __shared__ __align__(16) bf16_t lA[2 * BUFH];
  __shared__ __align__(16) bf16_t lB[2 * BUFH];
  v8f acc[2][4];
  const int m_block = blockIdx.x * BM;
  const int n_block = blockIdx.y * BN;
  gemm_mainloop(hsB, wB, D_, m_block, n_block, lA, lB, acc);

  const int lane = threadIdx.x & 31, wid = threadIdx.x >> 5;
  const int wm = wid & 3, wn = wid >> 2;
  const int mg0 = m_block + wm * 32 + ((lane >> 4) << 3);
  const int ng  = n_block + wn * 64 + (lane & 15);
  for (int i = 0; i < 2; i++) {
    for (int t = 0; t < 4; t++) {
      int n = ng + 16 * t;
      float bv = bias[n];
      int which = n >> 11;          // 0=q, 1=k, 2=v
      int rem = n & 2047;
      int head = rem >> 7, hd = rem & 127;
      bf16_t* dst = (which == 0) ? qB : ((which == 1) ? kB : vB);
      for (int j = 0; j < 8; j++) {
        int m = mg0 + 16 * i + j;
        int bb = m >> 11, ss = m & 2047;
        float v = acc[i][t][j] + bv;
        dst[(((size_t)(bb * H_ + head)) * S_ + ss) * HD_ + hd] = (bf16_t)v;
      }
    }
  }
}

// ---------------- Output projection: out = ctx @ out_w^T + b (fp32 out) ----------------
__global__ __launch_bounds__(256) void out_gemm_kernel(
    const bf16_t* __restrict__ ctxB, const bf16_t* __restrict__ owB,
    const float* __restrict__ bias, float* __restrict__ out) {
  __shared__ __align__(16) bf16_t lA[2 * BUFH];
  __shared__ __align__(16) bf16_t lB[2 * BUFH];
  v8f acc[2][4];
  const int m_block = blockIdx.x * BM;
  const int n_block = blockIdx.y * BN;
  gemm_mainloop(ctxB, owB, D_, m_block, n_block, lA, lB, acc);

  const int lane = threadIdx.x & 31, wid = threadIdx.x >> 5;
  const int wm = wid & 3, wn = wid >> 2;
  const int mg0 = m_block + wm * 32 + ((lane >> 4) << 3);
  const int ng  = n_block + wn * 64 + (lane & 15);
  for (int i = 0; i < 2; i++)
    for (int t = 0; t < 4; t++) {
      int n = ng + 16 * t;
      float bv = bias[n];
      for (int j = 0; j < 8; j++) {
        int m = mg0 + 16 * i + j;
        out[(size_t)m * D_ + n] = acc[i][t][j] + bv;
      }
    }
}

// ---------------- Flash attention: one (b,h) x 128 query rows per block ----------------
__global__ __launch_bounds__(256) void attn_kernel(
    const bf16_t* __restrict__ qB, const bf16_t* __restrict__ kB,
    const bf16_t* __restrict__ vB, bf16_t* __restrict__ ctxB) {
  __shared__ __align__(16) bf16_t Kt[KB_KEYS * KT_LD];   // [key][hd]
  __shared__ __align__(16) bf16_t Vt[HD_ * VT_LD];       // [hd][key] (transposed)
  __shared__ __align__(16) bf16_t Pb[8 * 16 * P_LD];     // per-wave P tiles

  const int tid = threadIdx.x, lane = tid & 31, wid = tid >> 5;
  const int bh = blockIdx.y;
  const int q0 = blockIdx.x * 128;
  const bf16_t* Qh = qB + (size_t)bh * S_ * HD_;
  const bf16_t* Kh = kB + (size_t)bh * S_ * HD_;
  const bf16_t* Vh = vB + (size_t)bh * S_ * HD_;

  // Q fragments for this wave's 16 rows (VGPR-resident for whole kernel)
  v16bf qf[4];
  for (int c = 0; c < 4; c++)
    qf[c] = ld_frag_a(Qh + (size_t)(q0 + wid * 16) * HD_ + 32 * c, HD_, lane);

  v8f z = {};
  v8f o[8];
  for (int n = 0; n < 8; n++) o[n] = z;
  float mi[8], li[8];
  for (int j = 0; j < 8; j++) { mi[j] = -INFINITY; li[j] = 0.0f; }
  const float scale = 0.08838834764831845f;   // 1/sqrt(128)

  const int srow = tid >> 2;            // key row 0..63
  const int soff = (tid & 3) << 5;      // hd halves: 0,32,64,96
  bf16_t* Pw = Pb + wid * 16 * P_LD;

  for (int kb = 0; kb < S_; kb += KB_KEYS) {
    const bf16_t* gK = Kh + (size_t)(kb + srow) * HD_ + soff;
    const bf16_t* gV = Vh + (size_t)(kb + srow) * HD_ + soff;

    // V rows to registers (needed for the transpose) -- overlap with prior compute
    union { uint4 q[4]; bf16_t h[32]; } vv;
    vv.q[0] = *(const uint4*)(gV);
    vv.q[1] = *(const uint4*)(gV + 8);
    vv.q[2] = *(const uint4*)(gV + 16);
    vv.q[3] = *(const uint4*)(gV + 24);

    __syncthreads();   // all waves done reading Kt/Vt from previous block

    // K tile: async global->LDS, no VGPR round-trip
    bf16_t* sK = Kt + srow * KT_LD + soff;
    async_copy_b128(gK,      sK);
    async_copy_b128(gK + 8,  sK + 8);
    async_copy_b128(gK + 16, sK + 16);
    async_copy_b128(gK + 24, sK + 24);

    // V transposed into LDS [hd][key]
#pragma unroll
    for (int iq = 0; iq < 32; iq++)
      Vt[(soff + iq) * VT_LD + srow] = vv.h[iq];

    WAIT_ASYNC(0);
    __syncthreads();

    // ---- scores: S = (Q @ K^T) * scale, 16x64 per wave ----
    v8f s[4];
    for (int t = 0; t < 4; t++) s[t] = z;
    for (int c = 0; c < 4; c++)
      for (int t = 0; t < 4; t++) {
        v16bf kf = ld_frag_b(Kt + (t * 16) * KT_LD + 32 * c, KT_LD, lane);
        s[t] = WMMA_BF16(qf[c], kf, s[t]);
      }
    for (int t = 0; t < 4; t++) s[t] = s[t] * scale;

    // ---- online softmax update (rows split across 16-lane halves) ----
    float alpha[8];
    for (int j = 0; j < 8; j++) {
      float mx = fmaxf(fmaxf(s[0][j], s[1][j]), fmaxf(s[2][j], s[3][j]));
      mx = fmaxf(mx, __shfl_xor(mx, 1, 32));
      mx = fmaxf(mx, __shfl_xor(mx, 2, 32));
      mx = fmaxf(mx, __shfl_xor(mx, 4, 32));
      mx = fmaxf(mx, __shfl_xor(mx, 8, 32));
      float mn = fmaxf(mi[j], mx);
      float a  = __expf(mi[j] - mn);
      float rs = 0.0f;
      for (int t = 0; t < 4; t++) {
        float p = __expf(s[t][j] - mn);
        s[t][j] = p;
        rs += p;
      }
      rs += __shfl_xor(rs, 1, 32);
      rs += __shfl_xor(rs, 2, 32);
      rs += __shfl_xor(rs, 4, 32);
      rs += __shfl_xor(rs, 8, 32);
      li[j] = li[j] * a + rs;
      mi[j] = mn;
      alpha[j] = a;
    }
    for (int n = 0; n < 8; n++)
      for (int j = 0; j < 8; j++)
        o[n][j] = o[n][j] * alpha[j];

    // ---- C-layout P -> A-layout via per-wave LDS bounce (bf16) ----
    {
      int rbase = (lane >> 4) << 3;
      int cidx  = lane & 15;
      for (int t = 0; t < 4; t++)
        for (int j = 0; j < 8; j++)
          Pw[(rbase + j) * P_LD + t * 16 + cidx] = (bf16_t)s[t][j];
    }

    // ---- O += P @ V ----
    for (int c = 0; c < 2; c++) {
      v16bf pf = ld_frag_a(Pw + 32 * c, P_LD, lane);
      for (int n = 0; n < 8; n++) {
        v16bf vf = ld_frag_b(Vt + (n * 16) * VT_LD + 32 * c, VT_LD, lane);
        o[n] = WMMA_BF16(pf, vf, o[n]);
      }
    }
  }

  // ---- epilogue: ctx[b][s][h][hd] = O / l ----
  const int b = bh >> 4, h = bh & 15;
  const int m0  = q0 + wid * 16 + ((lane >> 4) << 3);
  const int hd0 = lane & 15;
  for (int n = 0; n < 8; n++)
    for (int j = 0; j < 8; j++) {
      float v = o[n][j] / li[j];
      int m = m0 + j;
      int hd = n * 16 + hd0;
      ctxB[(((size_t)(b * S_ + m)) * H_ + h) * HD_ + hd] = (bf16_t)v;
    }
}

// ---------------- host launcher ----------------
extern "C" void kernel_launch(void* const* d_in, const int* in_sizes, int n_in,
                              void* d_out, int out_size, void* d_ws, size_t ws_size,
                              hipStream_t stream) {
  (void)in_sizes; (void)n_in; (void)out_size; (void)ws_size;
  const float* hs   = (const float*)d_in[0];
  const float* wqkv = (const float*)d_in[1];
  const float* bqkv = (const float*)d_in[2];
  const float* ow   = (const float*)d_in[3];
  const float* ob   = (const float*)d_in[4];
  float* out = (float*)d_out;

  char* ws = (char*)d_ws;
  size_t off = 0;
  auto alloc = [&](size_t bytes) {
    char* p = ws + off;
    off += (bytes + 255) & ~(size_t)255;
    return p;
  };
  bf16_t* hsB   = (bf16_t*)alloc((size_t)BS_ * D_ * 2);
  bf16_t* wqkvB = (bf16_t*)alloc((size_t)N3D_ * D_ * 2);
  bf16_t* owB   = (bf16_t*)alloc((size_t)D_ * D_ * 2);
  bf16_t* qB    = (bf16_t*)alloc((size_t)BS_ * D_ * 2);
  bf16_t* kB    = (bf16_t*)alloc((size_t)BS_ * D_ * 2);
  bf16_t* vB    = (bf16_t*)alloc((size_t)BS_ * D_ * 2);
  bf16_t* ctxB  = (bf16_t*)alloc((size_t)BS_ * D_ * 2);

  int n1 = BS_ * D_;
  cast_f32_to_bf16<<<n1 / 1024, 256, 0, stream>>>(hs, hsB, n1);
  int n2 = N3D_ * D_;
  cast_f32_to_bf16<<<n2 / 1024, 256, 0, stream>>>(wqkv, wqkvB, n2);
  int n3 = D_ * D_;
  cast_f32_to_bf16<<<n3 / 1024, 256, 0, stream>>>(ow, owB, n3);

  dim3 g1(BS_ / BM, N3D_ / BN);   // 32 x 48
  qkv_gemm_kernel<<<g1, 256, 0, stream>>>(hsB, wqkvB, bqkv, qB, kB, vB);

  dim3 g2(S_ / 128, B_ * H_);     // 16 x 32
  attn_kernel<<<g2, 256, 0, stream>>>(qB, kB, vB, ctxB);

  dim3 g3(BS_ / BM, D_ / BN);     // 32 x 16
  out_gemm_kernel<<<g3, 256, 0, stream>>>(ctxB, owB, ob, out);
}
